// DecoderLayer_24300924961529
// MI455X (gfx1250) — matrix operand order repaired and checked by reading, not compiled, not run on hardware
//
#include <hip/hip_runtime.h>
#include <hip/hip_bf16.h>
#include <stdint.h>

// ---------------------------------------------------------------------------
// Shapes (fixed by the reference): B=4 S=1024 H=1024 NH=16 NKV=4 HD=64
//                                  E=8 TOPK=2 F=2048, T = B*S = 4096
// ---------------------------------------------------------------------------
constexpr int TOK = 4096;
constexpr int HID = 1024;
constexpr int SEQ = 1024;
constexpr int NHq = 16;
constexpr int NKVq = 4;
constexpr int HDq = 64;
constexpr int Eq = 8;
constexpr int Fq = 2048;
constexpr int KVD = NKVq * HDq;   // 256

typedef __attribute__((ext_vector_type(16))) __bf16 v16bf;
typedef __attribute__((ext_vector_type(8)))  float  v8f;
typedef __attribute__((ext_vector_type(4)))  unsigned int u32x4;
typedef __attribute__((ext_vector_type(8)))  int i32x8;
typedef __attribute__((ext_vector_type(4)))  int i32x4;

union Frag { v16bf v; unsigned int u[8]; };

__device__ __forceinline__ unsigned short f2bfu(float f) {
  union { float f; unsigned int u; } v; v.f = f;
  unsigned int r = v.u + 0x7FFFu + ((v.u >> 16) & 1u);   // RNE
  return (unsigned short)(r >> 16);
}
__device__ __forceinline__ float bf2f(unsigned short u) {
  union { unsigned int u; float f; } v; v.u = (unsigned)u << 16; return v.f;
}
__device__ __forceinline__ v8f vzero() {
  v8f z;
#pragma unroll
  for (int i = 0; i < 8; ++i) z[i] = 0.0f;
  return z;
}
__device__ __forceinline__ v8f wmma_bf16(const Frag& a, const Frag& b, v8f c) {
  return __builtin_amdgcn_wmma_f32_16x16x32_bf16(false, a.v, false, b.v,
                                                 (short)0, c, false, false);
}

constexpr int BKP = 36;   // LDS K-stride (bf16 elems): 64B payload + 8B pad

// ---------------------------------------------------------------------------
// Tensor Data Mover: 2D strided tile (bf16 rows) -> LDS, with LDS padding.
// unitsX: row payload in 8-byte units. padIv: pad_interval code
// (3 -> pad after 64B, 4 -> pad after 128B); pad_amount fixed = 8 bytes.
// Issue from one wave; TENSORcnt completes in issue order per wave, so a
// double-buffered pipeline waits on "<= #descriptors of the next tile".
// ---------------------------------------------------------------------------
__device__ __forceinline__ unsigned lds_off(const void* p) {
  return (unsigned)(unsigned long long)(uintptr_t)p;
}
__device__ __forceinline__ void tdm_load_2d(unsigned ldsOff, const void* gaddr,
                                            unsigned unitsX, unsigned rows,
                                            unsigned strideUnits, unsigned padIv) {
  const unsigned long long ga = (unsigned long long)(uintptr_t)gaddr;
  u32x4 g0;
  g0[0] = 1u;                                       // count=1, user descriptor
  g0[1] = ldsOff;                                   // LDS byte address
  g0[2] = (unsigned)(ga & 0xFFFFFFFFu);             // global_addr[31:0]
  g0[3] = (unsigned)((ga >> 32) & 0x01FFFFFFu) | (2u << 30);  // [56:32] | type=2
  const unsigned td0 = 0x40000000u, td1 = 0x40000000u;        // no OOB clipping
  i32x8 g1;
  // data_size=3 (8B) | pad_enable | pad_interval=padIv | pad_amount=1 (8B)
  g1[0] = (int)((3u << 16) | (1u << 20) | (padIv << 22) | (1u << 25));
  g1[1] = (int)((td0 & 0xFFFFu) << 16);             // atomic_barrier=0 | td0.lo
  g1[2] = (int)((td0 >> 16) | ((td1 & 0xFFFFu) << 16));
  g1[3] = (int)((td1 >> 16) | (unitsX << 16));      // td1.hi | tile_dim0
  g1[4] = (int)(rows & 0xFFFFu);                    // tile_dim1 | tile_dim2=0
  g1[5] = (int)strideUnits;                         // tensor_dim0_stride[31:0]
  g1[6] = 0;
  g1[7] = 0;
  i32x4 gz = {0, 0, 0, 0};
#if __clang_major__ >= 23
  i32x8 gz8 = {0, 0, 0, 0, 0, 0, 0, 0};
  __builtin_amdgcn_tensor_load_to_lds(g0, g1, gz, gz, gz8, 0);
#else
  __builtin_amdgcn_tensor_load_to_lds(g0, g1, gz, gz, 0);
#endif
}

// A fragment (16x32 bf16, MxK): lane m=l&15; lanes<16 hold K 0..15,
// lanes>=16 hold K 16..31; K pairs contiguous -> 32-bit LDS loads.
__device__ __forceinline__ Frag load_a_frag(const unsigned short* s, int lane) {
  const int m = lane & 15, ks = (lane >> 4) * 16;
  const unsigned int* row = (const unsigned int*)(s + m * BKP);
  Frag f;
#pragma unroll
  for (int p = 0; p < 4; ++p) {
    f.u[p]     = row[(ks >> 1) + p];
    f.u[4 + p] = row[(ks >> 1) + 4 + p];
  }
  return f;
}
// B fragment (32x16 bf16, KxN) from K-major LDS tile: lane n=l&15;
// lanes<16 hold K 0..15, lanes>=16 hold K 16..31 (contiguous).
__device__ __forceinline__ Frag load_b_frag(const unsigned short* s, int lane) {
  const int n = lane & 15, ks = (lane >> 4) * 16;
  const unsigned int* row = (const unsigned int*)(s + n * BKP);
  Frag f;
#pragma unroll
  for (int p = 0; p < 8; ++p) f.u[p] = row[(ks >> 1) + p];
  return f;
}

// ---------------------------------------------------------------------------
// One-time weight convert + transpose: f32 [K,N] (batched) -> bf16 [N,K]
// ---------------------------------------------------------------------------
__global__ __launch_bounds__(256)
void wcvt_t_k(const float* __restrict__ In, unsigned short* __restrict__ Out,
              int K, int N) {
  const float* in = In + (size_t)blockIdx.z * K * N;
  unsigned short* out = Out + (size_t)blockIdx.z * K * N;
  __shared__ unsigned short t[32][33];
  const int k0 = blockIdx.y * 32, n0 = blockIdx.x * 32;
  const int tx = threadIdx.x & 31, ty = threadIdx.x >> 5;
#pragma unroll
  for (int i = 0; i < 32; i += 8)
    t[ty + i][tx] = f2bfu(in[(size_t)(k0 + ty + i) * N + n0 + tx]);
  __syncthreads();
#pragma unroll
  for (int i = 0; i < 32; i += 8)
    out[(size_t)(n0 + ty + i) * K + k0 + tx] = t[tx][ty + i];
}

// ---------------------------------------------------------------------------
// RMSNorm: one block per token. f32 in, bf16 out.
// ---------------------------------------------------------------------------
__global__ __launch_bounds__(256)
void rmsnorm_k(const float* __restrict__ X, const float* __restrict__ W,
               unsigned short* __restrict__ Y) {
  const int t = blockIdx.x, tid = threadIdx.x;
  const float* xr = X + (size_t)t * HID;
  float ss = 0.0f;
#pragma unroll
  for (int j = tid; j < HID; j += 256) { float v = xr[j]; ss += v * v; }
  __shared__ float red[256];
  red[tid] = ss;
  __syncthreads();
  for (int s = 128; s > 0; s >>= 1) {
    if (tid < s) red[tid] += red[tid + s];
    __syncthreads();
  }
  const float rs = rsqrtf(red[0] * (1.0f / HID) + 1e-6f);
#pragma unroll
  for (int j = tid; j < HID; j += 256)
    Y[(size_t)t * HID + j] = f2bfu(W[j] * xr[j] * rs);
}

// ---------------------------------------------------------------------------
// WMMA GEMM: C[M,N] = A[M,K] * Bt[N,K]^T, bf16 A/B, f32 acc.
// Double-buffered TDM pipeline: wave 0 issues tile i+1, waits tensorcnt<=2
// (tile i resident), all waves compute tile i while tile i+1 streams in.
// ---------------------------------------------------------------------------
template<int BM, int BN, int WM, int WN, bool RES, bool OUTBF>
__global__ __launch_bounds__(256)
void gemm_bf16(const unsigned short* __restrict__ A,
               const unsigned short* __restrict__ Bt,
               void* __restrict__ Cout, const float* __restrict__ Rsd,
               int M, int N, int K) {
  constexpr int WTM = BM / WM, WTN = BN / WN;
  constexpr int IT = WTM / 16, JT = WTN / 16;
  __shared__ unsigned short aS[2][BM * BKP];
  __shared__ unsigned short bS[2][BN * BKP];
  const int tid = threadIdx.x, lane = tid & 31, wid = tid >> 5;
  const int wm = wid / WN, wn = wid % WN;
  const int m0 = blockIdx.y * BM, n0 = blockIdx.x * BN;
  const unsigned ku = (unsigned)K >> 2;

  v8f acc[IT][JT];
#pragma unroll
  for (int i = 0; i < IT; ++i)
#pragma unroll
    for (int j = 0; j < JT; ++j) acc[i][j] = vzero();

  if (tid < 32) {                               // prologue: tile 0
    tdm_load_2d(lds_off(aS[0]), A + (size_t)m0 * K, 8, BM, ku, 3);
    tdm_load_2d(lds_off(bS[0]), Bt + (size_t)n0 * K, 8, BN, ku, 3);
  }
  int idx = 0;
  for (int k0 = 0; k0 < K; k0 += 32, idx ^= 1) {
    const bool nxt = (k0 + 32) < K;
    __syncthreads();                            // buf[idx^1] free
    if (tid < 32) {
      if (nxt) {
        tdm_load_2d(lds_off(aS[idx ^ 1]), A + (size_t)m0 * K + k0 + 32, 8, BM, ku, 3);
        tdm_load_2d(lds_off(bS[idx ^ 1]), Bt + (size_t)n0 * K + k0 + 32, 8, BN, ku, 3);
        __builtin_amdgcn_s_wait_tensorcnt(2);   // tile i done; i+1 in flight
      } else {
        __builtin_amdgcn_s_wait_tensorcnt(0);
      }
    }
    __syncthreads();                            // buf[idx] visible
    Frag af[IT], bfr[JT];
#pragma unroll
    for (int i = 0; i < IT; ++i)
      af[i] = load_a_frag(aS[idx] + (wm * WTM + i * 16) * BKP, lane);
#pragma unroll
    for (int j = 0; j < JT; ++j)
      bfr[j] = load_b_frag(bS[idx] + (wn * WTN + j * 16) * BKP, lane);
#pragma unroll
    for (int i = 0; i < IT; ++i)
#pragma unroll
      for (int j = 0; j < JT; ++j) acc[i][j] = wmma_bf16(af[i], bfr[j], acc[i][j]);
  }
#pragma unroll
  for (int i = 0; i < IT; ++i)
#pragma unroll
    for (int j = 0; j < JT; ++j)
#pragma unroll
      for (int p = 0; p < 8; ++p) {
        const int row = m0 + wm * WTM + i * 16 + p + 8 * (lane >> 4);
        const int col = n0 + wn * WTN + j * 16 + (lane & 15);
        float v = acc[i][j][p];
        if constexpr (OUTBF) {
          ((unsigned short*)Cout)[(size_t)row * N + col] = f2bfu(v);
        } else {
          if constexpr (RES) v += Rsd[(size_t)row * N + col];
          ((float*)Cout)[(size_t)row * N + col] = v;
        }
      }
}

// ---------------------------------------------------------------------------
// Flash attention, causal, GQA (16 q-heads over 4 kv-heads), all-bf16 I/O.
// K chunks double-buffered via TDM; V transpose-staged into the inactive
// buffer after compute. Grid: (SEQ/128, B*NH); wave owns 16 queries.
// ---------------------------------------------------------------------------
__global__ __launch_bounds__(256)
void flash_attn(const unsigned short* __restrict__ Qb,
                const unsigned short* __restrict__ Kb,
                const unsigned short* __restrict__ Vb,
                unsigned short* __restrict__ Ob) {
  const int b = blockIdx.y >> 4;
  const int h = blockIdx.y & 15;
  const int kvh = h >> 2;
  const int q0 = blockIdx.x * 128;
  const int tid = threadIdx.x, lane = tid & 31, wid = tid >> 5;
  const int qw = q0 + wid * 16;

  __shared__ unsigned short kS[2][32 * 68];   // [key][d], TDM (128B+8B pad)
  __shared__ unsigned short vS[2][64 * 36];   // [d][key], manual transpose
  __shared__ unsigned short pS[8 * 16 * 36];  // per-wave 16x32 P tile

  auto stageV = [&](int kk0, unsigned short* dst) {
#pragma unroll
    for (int it = 0; it < 2; ++it) {
      int u = (it * 256 + tid) * 4;
      int key = u >> 6, d = u & 63;
      const uint2 w = *(const uint2*)(Vb + (size_t)(b * SEQ + kk0 + key) * KVD +
                                      kvh * HDq + d);
      dst[(d + 0) * 36 + key] = (unsigned short)(w.x & 0xFFFFu);
      dst[(d + 1) * 36 + key] = (unsigned short)(w.x >> 16);
      dst[(d + 2) * 36 + key] = (unsigned short)(w.y & 0xFFFFu);
      dst[(d + 3) * 36 + key] = (unsigned short)(w.y >> 16);
    }
  };

  Frag qa[2];                                  // Q A-fragments, d 0..31 / 32..63
  {
    const int m = qw + (lane & 15);
    const unsigned short* qrow = Qb + (size_t)(b * SEQ + m) * HID + h * HDq;
    const int ks = (lane >> 4) * 16;
#pragma unroll
    for (int hh = 0; hh < 2; ++hh)
#pragma unroll
      for (int p = 0; p < 8; ++p) {
        int k = hh * 32 + ks + 2 * p;
        qa[hh].u[p] = *(const unsigned int*)(qrow + k);
      }
  }

  float mrun[8], lrun[8];
  v8f accO[4];
#pragma unroll
  for (int p = 0; p < 8; ++p) { mrun[p] = -1e30f; lrun[p] = 0.0f; }
#pragma unroll
  for (int j = 0; j < 4; ++j) accO[j] = vzero();

  const int nCh = q0 / 32 + 4;                 // keys needed: <= q0+127
  stageV(0, vS[0]);                            // prologue: chunk 0
  if (tid < 32)
    tdm_load_2d(lds_off(kS[0]), Kb + (size_t)(b * SEQ) * KVD + kvh * HDq,
                16, 32, KVD >> 2, 4);
  int idx = 0;
  for (int ch = 0; ch < nCh; ++ch, idx ^= 1) {
    const int kk0 = ch * 32;
    const bool nxt = (ch + 1) < nCh;
    __syncthreads();
    if (tid < 32) {
      if (nxt) {
        tdm_load_2d(lds_off(kS[idx ^ 1]),
                    Kb + (size_t)(b * SEQ + kk0 + 32) * KVD + kvh * HDq,
                    16, 32, KVD >> 2, 4);
        __builtin_amdgcn_s_wait_tensorcnt(1);
      } else {
        __builtin_amdgcn_s_wait_tensorcnt(0);
      }
    }
    __syncthreads();
    if (kk0 <= qw + 15) {                      // not fully masked for this wave
      v8f sc[2];
#pragma unroll
      for (int t = 0; t < 2; ++t) {            // two 16-key score tiles
        Frag b0, b1;
        {
          const int n = t * 16 + (lane & 15);
          const int db = (lane >> 4) * 16;
          const unsigned int* r0 = (const unsigned int*)(kS[idx] + n * 68 + db);
          const unsigned int* r1 = (const unsigned int*)(kS[idx] + n * 68 + 32 + db);
#pragma unroll
          for (int p = 0; p < 8; ++p) { b0.u[p] = r0[p]; b1.u[p] = r1[p]; }
        }
        v8f s = vzero();
        s = wmma_bf16(qa[0], b0, s);
        s = wmma_bf16(qa[1], b1, s);
#pragma unroll
        for (int p = 0; p < 8; ++p) {          // scale (1/sqrt(64)) + causal
          int row = qw + p + 8 * (lane >> 4);
          int key = kk0 + t * 16 + (lane & 15);
          s[p] = (key > row) ? -1e30f : s[p] * 0.125f;
        }
        sc[t] = s;
      }
      // online softmax, per row (rows replicated across each 16-lane group)
#pragma unroll
      for (int p = 0; p < 8; ++p) {
        float rv = fmaxf(sc[0][p], sc[1][p]);
        rv = fmaxf(rv, __shfl_xor(rv, 1, 32));
        rv = fmaxf(rv, __shfl_xor(rv, 2, 32));
        rv = fmaxf(rv, __shfl_xor(rv, 4, 32));
        rv = fmaxf(rv, __shfl_xor(rv, 8, 32));
        const float mnew = fmaxf(mrun[p], rv);
        const float scale = __expf(mrun[p] - mnew);
        const float p0 = __expf(sc[0][p] - mnew);
        const float p1 = __expf(sc[1][p] - mnew);
        float rs = p0 + p1;
        rs += __shfl_xor(rs, 1, 32);
        rs += __shfl_xor(rs, 2, 32);
        rs += __shfl_xor(rs, 4, 32);
        rs += __shfl_xor(rs, 8, 32);
        lrun[p] = lrun[p] * scale + rs;
        mrun[p] = mnew;
#pragma unroll
        for (int j = 0; j < 4; ++j) accO[j][p] *= scale;
        unsigned short* pr = pS + wid * 576 + (p + 8 * (lane >> 4)) * 36;
        pr[(lane & 15)]      = f2bfu(p0);
        pr[16 + (lane & 15)] = f2bfu(p1);
      }
      // P(16x32) @ V(32x64)
      Frag pa = load_a_frag(pS + wid * 576, lane);
#pragma unroll
      for (int j = 0; j < 4; ++j) {
        Frag vf;
        {
          const int n = j * 16 + (lane & 15);
          const int kb = (lane >> 4) * 16;
          const unsigned int* rp = (const unsigned int*)(vS[idx] + n * 36 + kb);
#pragma unroll
          for (int p = 0; p < 8; ++p) vf.u[p] = rp[p];
        }
        accO[j] = wmma_bf16(pa, vf, accO[j]);
      }
    }
    if (nxt) stageV(kk0 + 32, vS[idx ^ 1]);    // overlap with compute drain
  }
#pragma unroll
  for (int j = 0; j < 4; ++j)
#pragma unroll
    for (int p = 0; p < 8; ++p) {
      const int row = qw + p + 8 * (lane >> 4);
      const int col = h * HDq + j * 16 + (lane & 15);
      Ob[(size_t)(b * SEQ + row) * HID + col] = f2bfu(accO[j][p] / lrun[p]);
    }
}

// ---------------------------------------------------------------------------
// MoE routing
// ---------------------------------------------------------------------------
__global__ void zero_k(int* cnt) { if (threadIdx.x < Eq) cnt[threadIdx.x] = 0; }

__global__ __launch_bounds__(128)
void router_k(const unsigned short* __restrict__ X2,
              const float* __restrict__ Wgate, int* __restrict__ eids,
              float* __restrict__ wts, int* __restrict__ cnt) {
  const int t = blockIdx.x, tid = threadIdx.x;
  const unsigned short* xr = X2 + (size_t)t * HID;
  float a[Eq];
#pragma unroll
  for (int e = 0; e < Eq; ++e) a[e] = 0.0f;
  for (int j = tid; j < HID; j += 128) {
    const float xv = bf2f(xr[j]);
    const float* wr = Wgate + (size_t)j * Eq;
#pragma unroll
    for (int e = 0; e < Eq; ++e) a[e] += xv * wr[e];
  }
  __shared__ float red[128 * Eq];
#pragma unroll
  for (int e = 0; e < Eq; ++e) red[tid * Eq + e] = a[e];
  __syncthreads();
  for (int s = 64; s > 0; s >>= 1) {
    if (tid < s)
#pragma unroll
      for (int e = 0; e < Eq; ++e) red[tid * Eq + e] += red[(tid + s) * Eq + e];
    __syncthreads();
  }
  if (tid == 0) {
    float pr[Eq];
    float mx = red[0];
#pragma unroll
    for (int e = 1; e < Eq; ++e) mx = fmaxf(mx, red[e]);
    float sum = 0.0f;
#pragma unroll
    for (int e = 0; e < Eq; ++e) { pr[e] = __expf(red[e] - mx); sum += pr[e]; }
    const float inv = 1.0f / sum;
#pragma unroll
    for (int e = 0; e < Eq; ++e) pr[e] *= inv;
    int i0 = 0;
#pragma unroll
    for (int e = 1; e < Eq; ++e) if (pr[e] > pr[i0]) i0 = e;
    int i1 = (i0 == 0) ? 1 : 0;
#pragma unroll
    for (int e = 0; e < Eq; ++e) if (e != i0 && pr[e] > pr[i1]) i1 = e;
    eids[t * 2 + 0] = i0; wts[t * 2 + 0] = pr[i0];
    eids[t * 2 + 1] = i1; wts[t * 2 + 1] = pr[i1];
    atomicAdd(&cnt[i0], 1);
    atomicAdd(&cnt[i1], 1);
  }
}

__global__ void offsets_k(const int* cnt, int* offs, int* curs) {
  if (threadIdx.x == 0) {
    int acc = 0;
    for (int e = 0; e < Eq; ++e) { offs[e] = acc; curs[e] = acc; acc += cnt[e]; }
  }
}

__global__ __launch_bounds__(256)
void scatter_k(const int* __restrict__ eids, const float* __restrict__ wts,
               int* __restrict__ curs, int* __restrict__ list,
               float* __restrict__ wlist, int* __restrict__ pos) {
  const int t = blockIdx.x * 256 + threadIdx.x;
#pragma unroll
  for (int s = 0; s < 2; ++s) {
    const int e = eids[t * 2 + s];
    const int r = atomicAdd(&curs[e], 1);
    list[r] = t;
    wlist[r] = wts[t * 2 + s];
    pos[t * 2 + s] = r;
  }
}

// ---------------------------------------------------------------------------
// MoE up: act[r,f] = silu(x2[tok] @ Wg_e) * (x2[tok] @ Wu_e) * w   (bf16)
// Grid: (F/64, 32 token tiles, E). Wg/Wu tiles double-buffered via TDM;
// the gathered-A copy writes the inactive buffer after compute.
// ---------------------------------------------------------------------------
__global__ __launch_bounds__(256)
void moe_up_k(const unsigned short* __restrict__ X2,
              const unsigned short* __restrict__ Wgt,
              const unsigned short* __restrict__ Wut,
              const int* __restrict__ list, const float* __restrict__ wlist,
              const int* __restrict__ cnt, const int* __restrict__ offs,
              unsigned short* __restrict__ act) {
  const int e = blockIdx.z;
  const int nTok = cnt[e];
  const int r0 = blockIdx.y * 128;
  if (r0 >= nTok) return;
  const int base = offs[e];
  const int n0 = blockIdx.x * 64;
  const unsigned short* Bg = Wgt + (size_t)e * Fq * HID;   // [F,H] bf16
  const unsigned short* Bu = Wut + (size_t)e * Fq * HID;

  __shared__ int rowTok[128];
  __shared__ float rowW[128];
  __shared__ unsigned short aS[2][128 * BKP];
  __shared__ unsigned short gS[2][64 * BKP], uS[2][64 * BKP];
  const int tid = threadIdx.x, lane = tid & 31, wid = tid >> 5;
  const int wm = wid >> 1, wn = wid & 1;
  if (tid < 128) {
    int r = r0 + tid; if (r >= nTok) r = nTok - 1;
    rowTok[tid] = list[base + r];
    rowW[tid] = wlist[base + r];
  }
  __syncthreads();                              // rowTok visible

  auto stageA = [&](int k0, unsigned short* dst) {
#pragma unroll
    for (int it = 0; it < 4; ++it) {
      int u = (it * 256 + tid) * 4;
      int r = u >> 5, c = u & 31;
      *(uint2*)(dst + r * BKP + c) =
          *(const uint2*)(X2 + (size_t)rowTok[r] * HID + k0 + c);
    }
  };

  v8f accG[2][2], accU[2][2];
#pragma unroll
  for (int i = 0; i < 2; ++i)
#pragma unroll
    for (int j = 0; j < 2; ++j) { accG[i][j] = vzero(); accU[i][j] = vzero(); }

  stageA(0, aS[0]);                             // prologue: tile 0
  if (tid < 32) {
    tdm_load_2d(lds_off(gS[0]), Bg + (size_t)n0 * HID, 8, 64, HID >> 2, 3);
    tdm_load_2d(lds_off(uS[0]), Bu + (size_t)n0 * HID, 8, 64, HID >> 2, 3);
  }
  int idx = 0;
  for (int k0 = 0; k0 < HID; k0 += 32, idx ^= 1) {
    const bool nxt = (k0 + 32) < HID;
    __syncthreads();
    if (tid < 32) {
      if (nxt) {
        tdm_load_2d(lds_off(gS[idx ^ 1]), Bg + (size_t)n0 * HID + k0 + 32,
                    8, 64, HID >> 2, 3);
        tdm_load_2d(lds_off(uS[idx ^ 1]), Bu + (size_t)n0 * HID + k0 + 32,
                    8, 64, HID >> 2, 3);
        __builtin_amdgcn_s_wait_tensorcnt(2);
      } else {
        __builtin_amdgcn_s_wait_tensorcnt(0);
      }
    }
    __syncthreads();
    Frag af[2], gb[2], ub[2];
#pragma unroll
    for (int i = 0; i < 2; ++i)
      af[i] = load_a_frag(aS[idx] + (wm * 32 + i * 16) * BKP, lane);
#pragma unroll
    for (int j = 0; j < 2; ++j) {
      gb[j] = load_b_frag(gS[idx] + (wn * 32 + j * 16) * BKP, lane);
      ub[j] = load_b_frag(uS[idx] + (wn * 32 + j * 16) * BKP, lane);
    }
#pragma unroll
    for (int i = 0; i < 2; ++i)
#pragma unroll
      for (int j = 0; j < 2; ++j) {
        accG[i][j] = wmma_bf16(af[i], gb[j], accG[i][j]);
        accU[i][j] = wmma_bf16(af[i], ub[j], accU[i][j]);
      }
    if (nxt) stageA(k0 + 32, aS[idx ^ 1]);      // overlap with WMMA drain
  }
#pragma unroll
  for (int i = 0; i < 2; ++i)
#pragma unroll
    for (int j = 0; j < 2; ++j)
#pragma unroll
      for (int p = 0; p < 8; ++p) {
        const int lr = wm * 32 + i * 16 + p + 8 * (lane >> 4);
        if (r0 + lr < nTok) {
          const float g = accG[i][j][p], u = accU[i][j][p];
          const float a = (g / (1.0f + __expf(-g))) * u * rowW[lr];
          const int col = n0 + wn * 32 + j * 16 + (lane & 15);
          act[(size_t)(base + r0 + lr) * Fq + col] = f2bfu(a);
        }
      }
}

// ---------------------------------------------------------------------------
// MoE down: dn[r,h] = act[r,:] @ Wd_e.  A and B double-buffered via TDM.
// Grid: (H/128, 32 token tiles, E).
// ---------------------------------------------------------------------------
__global__ __launch_bounds__(256)
void moe_down_k(const unsigned short* __restrict__ act,
                const unsigned short* __restrict__ Wdt,
                const int* __restrict__ cnt, const int* __restrict__ offs,
                float* __restrict__ dn) {
  const int e = blockIdx.z;
  const int nTok = cnt[e];
  const int r0 = blockIdx.y * 128;
  if (r0 >= nTok) return;
  const int base = offs[e];
  const int n0 = blockIdx.x * 128;
  const unsigned short* Bd = Wdt + (size_t)e * HID * Fq;   // [H,F] bf16
  const unsigned short* Arow = act + (size_t)(base + r0) * Fq;

  __shared__ unsigned short aS[2][128 * BKP], bS[2][128 * BKP];
  const int tid = threadIdx.x, lane = tid & 31, wid = tid >> 5;
  const int wm = wid >> 1, wn = wid & 1;
  v8f acc[2][4];
#pragma unroll
  for (int i = 0; i < 2; ++i)
#pragma unroll
    for (int j = 0; j < 4; ++j) acc[i][j] = vzero();

  if (tid < 32) {
    tdm_load_2d(lds_off(aS[0]), Arow, 8, 128, Fq >> 2, 3);
    tdm_load_2d(lds_off(bS[0]), Bd + (size_t)n0 * Fq, 8, 128, Fq >> 2, 3);
  }
  int idx = 0;
  for (int k0 = 0; k0 < Fq; k0 += 32, idx ^= 1) {
    const bool nxt = (k0 + 32) < Fq;
    __syncthreads();
    if (tid < 32) {
      if (nxt) {
        tdm_load_2d(lds_off(aS[idx ^ 1]), Arow + k0 + 32, 8, 128, Fq >> 2, 3);
        tdm_load_2d(lds_off(bS[idx ^ 1]), Bd + (size_t)n0 * Fq + k0 + 32,
                    8, 128, Fq >> 2, 3);
        __builtin_amdgcn_s_wait_tensorcnt(2);
      } else {
        __builtin_amdgcn_s_wait_tensorcnt(0);
      }
    }
    __syncthreads();
    Frag af[2], bfr[4];
#pragma unroll
    for (int i = 0; i < 2; ++i)
      af[i] = load_a_frag(aS[idx] + (wm * 32 + i * 16) * BKP, lane);
#pragma unroll
    for (int j = 0; j < 4; ++j)
      bfr[j] = load_b_frag(bS[idx] + (wn * 64 + j * 16) * BKP, lane);
#pragma unroll
    for (int i = 0; i < 2; ++i)
#pragma unroll
      for (int j = 0; j < 4; ++j) acc[i][j] = wmma_bf16(af[i], bfr[j], acc[i][j]);
  }
#pragma unroll
  for (int i = 0; i < 2; ++i)
#pragma unroll
    for (int j = 0; j < 4; ++j)
#pragma unroll
      for (int p = 0; p < 8; ++p) {
        const int lr = wm * 32 + i * 16 + p + 8 * (lane >> 4);
        if (r0 + lr < nTok) {
          const int col = n0 + wn * 64 + j * 16 + (lane & 15);
          dn[(size_t)(base + r0 + lr) * HID + col] = acc[i][j][p];
        }
      }
}

// out[t] = h[t] + dn[pos0(t)] + dn[pos1(t)]  (routing weights folded in act)
__global__ __launch_bounds__(256)
void final_k(const float* __restrict__ H, const float* __restrict__ Dn,
             const int* __restrict__ pos, float* __restrict__ Out) {
  const int t = blockIdx.x, tid = threadIdx.x;
  const float* d0 = Dn + (size_t)pos[t * 2 + 0] * HID;
  const float* d1 = Dn + (size_t)pos[t * 2 + 1] * HID;
#pragma unroll
  for (int j = tid; j < HID; j += 256)
    Out[(size_t)t * HID + j] = H[(size_t)t * HID + j] + d0[j] + d1[j];
}

// ---------------------------------------------------------------------------
// Host launcher
// ---------------------------------------------------------------------------
extern "C" void kernel_launch(void* const* d_in, const int* in_sizes, int n_in,
                              void* d_out, int out_size, void* d_ws, size_t ws_size,
                              hipStream_t stream) {
  (void)in_sizes; (void)n_in; (void)out_size; (void)ws_size;
  const float* x     = (const float*)d_in[0];
  // d_in[1] = attention_mask (causal; applied analytically)
  const float* w_ln1 = (const float*)d_in[2];
  const float* w_ln2 = (const float*)d_in[3];
  const float* Wq    = (const float*)d_in[4];
  const float* Wk    = (const float*)d_in[5];
  const float* Wv    = (const float*)d_in[6];
  const float* Wo    = (const float*)d_in[7];
  const float* Wgate = (const float*)d_in[8];
  const float* Wg    = (const float*)d_in[9];
  const float* Wu    = (const float*)d_in[10];
  const float* Wd    = (const float*)d_in[11];
  float* out = (float*)d_out;

  char* wp = (char*)d_ws;
  auto alloc = [&](size_t n) -> char* {
    char* p = wp; wp += (n + 255) & ~(size_t)255; return p;
  };
  typedef unsigned short us;
  const size_t TH = (size_t)TOK * HID;
  us*    x1  = (us*)alloc(TH * 2);
  us*    qb  = (us*)alloc(TH * 2);
  us*    kb  = (us*)alloc((size_t)TOK * KVD * 2);
  us*    vb  = (us*)alloc((size_t)TOK * KVD * 2);
  us*    ao  = (us*)alloc(TH * 2);
  float* hb  = (float*)alloc(TH * 4);
  us*    x2  = (us*)alloc(TH * 2);
  us*    act = (us*)alloc(((size_t)2 * TOK + 128) * Fq * 2);
  float* dn  = (float*)alloc((size_t)2 * TOK * HID * 4);
  us*    Wqt = (us*)alloc((size_t)HID * HID * 2);
  us*    Wkt = (us*)alloc((size_t)HID * KVD * 2);
  us*    Wvt = (us*)alloc((size_t)HID * KVD * 2);
  us*    Wot = (us*)alloc((size_t)HID * HID * 2);
  us*    Wgt = (us*)alloc((size_t)Eq * HID * Fq * 2);
  us*    Wut = (us*)alloc((size_t)Eq * HID * Fq * 2);
  us*    Wdt = (us*)alloc((size_t)Eq * Fq * HID * 2);
  int*   eids = (int*)alloc((size_t)TOK * 2 * 4);
  float* wts  = (float*)alloc((size_t)TOK * 2 * 4);
  int*   pos  = (int*)alloc((size_t)TOK * 2 * 4);
  int*   list = (int*)alloc((size_t)2 * TOK * 4);
  float* wlist= (float*)alloc((size_t)2 * TOK * 4);
  int*   cnt  = (int*)alloc(64);
  int*   offs = (int*)alloc(64);
  int*   curs = (int*)alloc(64);

  // one-time weight convert+transpose: f32 [K,N] -> bf16 [N,K]
  wcvt_t_k<<<dim3(HID/32, HID/32, 1), 256, 0, stream>>>(Wq, Wqt, HID, HID);
  wcvt_t_k<<<dim3(KVD/32, HID/32, 1), 256, 0, stream>>>(Wk, Wkt, HID, KVD);
  wcvt_t_k<<<dim3(KVD/32, HID/32, 1), 256, 0, stream>>>(Wv, Wvt, HID, KVD);
  wcvt_t_k<<<dim3(HID/32, HID/32, 1), 256, 0, stream>>>(Wo, Wot, HID, HID);
  wcvt_t_k<<<dim3(Fq/32, HID/32, Eq), 256, 0, stream>>>(Wg, Wgt, HID, Fq);
  wcvt_t_k<<<dim3(Fq/32, HID/32, Eq), 256, 0, stream>>>(Wu, Wut, HID, Fq);
  wcvt_t_k<<<dim3(HID/32, Fq/32, Eq), 256, 0, stream>>>(Wd, Wdt, Fq, HID);

  rmsnorm_k<<<TOK, 256, 0, stream>>>(x, w_ln1, x1);
  gemm_bf16<128,128,4,2,false,true><<<dim3(HID/128, TOK/128), 256, 0, stream>>>(
      x1, Wqt, qb, nullptr, TOK, HID, HID);
  gemm_bf16<128,128,4,2,false,true><<<dim3(KVD/128, TOK/128), 256, 0, stream>>>(
      x1, Wkt, kb, nullptr, TOK, KVD, HID);
  gemm_bf16<128,128,4,2,false,true><<<dim3(KVD/128, TOK/128), 256, 0, stream>>>(
      x1, Wvt, vb, nullptr, TOK, KVD, HID);
  flash_attn<<<dim3(SEQ/128, (TOK/SEQ) * NHq), 256, 0, stream>>>(qb, kb, vb, ao);
  gemm_bf16<128,128,4,2,true,false><<<dim3(HID/128, TOK/128), 256, 0, stream>>>(
      ao, Wot, hb, x, TOK, HID, HID);
  rmsnorm_k<<<TOK, 256, 0, stream>>>(hb, w_ln2, x2);
  zero_k<<<1, 32, 0, stream>>>(cnt);
  router_k<<<TOK, 128, 0, stream>>>(x2, Wgate, eids, wts, cnt);
  offsets_k<<<1, 1, 0, stream>>>(cnt, offs, curs);
  scatter_k<<<TOK / 256, 256, 0, stream>>>(eids, wts, curs, list, wlist, pos);
  moe_up_k<<<dim3(Fq/64, TOK/128, Eq), 256, 0, stream>>>(
      x2, Wgt, Wut, list, wlist, cnt, offs, act);
  moe_down_k<<<dim3(HID/128, TOK/128, Eq), 256, 0, stream>>>(act, Wdt, cnt, offs, dn);
  final_k<<<TOK, 256, 0, stream>>>(hb, dn, pos, out);
}
